// SpatialCrossAttentionFusionGate_62715112456649
// MI455X (gfx1250) — matrix operand order repaired and checked by reading, not compiled, not run on hardware
//
#include <hip/hip_runtime.h>
#include <cstdint>

typedef _Float16 h16;
typedef __attribute__((ext_vector_type(8)))  _Float16 v8h;
typedef __attribute__((ext_vector_type(16))) _Float16 v16h;
typedef __attribute__((ext_vector_type(8)))  float    v8f;
typedef __attribute__((ext_vector_type(4)))  unsigned int u32x4;
typedef __attribute__((ext_vector_type(4)))  int i32x4;
typedef __attribute__((ext_vector_type(8)))  int i32x8;

#define DEVFN static __device__ __forceinline__

namespace cfg {
constexpr int B = 32, C = 256, T = 64, V = 25, H = 8, D = 32;
constexpr int C3 = 3 * C;            // 768
constexpr int TV = T * V;            // 1600
constexpr int XP = 264;              // pitch (halves) of X / Fused tiles (16B aligned)
constexpr int QP = 40;               // pitch (halves) of q/k/v and P tiles (conflict-free)
constexpr int YP = 257;              // pitch (floats) of Y buffer (conflict-free col reads)
// LDS layout (bytes)
constexpr int SZ_X    = 32 * XP * 2;              // 16896
constexpr int SZ_TILE = 32 * QP * 2;              // 2560
constexpr int OFF_X0  = 0;
constexpr int OFF_X1  = OFF_X0 + SZ_X;
constexpr int OFF_QKV = OFF_X1 + SZ_X;            // 33792  (48 tiles: [br][q/k/v][head])
constexpr int OFF_P   = OFF_QKV + 48 * SZ_TILE;   // 156672 (one P tile per wave)
constexpr int OFF_F   = OFF_P + 8 * SZ_TILE;      // 177152
constexpr int OFF_Y   = OFF_F + SZ_X;             // 194048
constexpr int SMEM    = OFF_Y + 32 * YP * 4;      // 226944
// TDM staging (f32, [c][v] compact 256x25): overlays P/F/Y regions -- those are
// only written after the staging data has been consumed (phase 1 barrier).
constexpr int OFF_STG = OFF_P;                    // 2 x 25600 B = 51200, ends 207872 < SMEM
constexpr int SZ_STG  = C * V * 4;                // 25600
}

DEVFN v8f wmma_f16(v16h a, v16h b, v8f c) {
  return __builtin_amdgcn_wmma_f32_16x16x32_f16(false, a, false, b, (short)0, c,
                                                false, false);
}

// A fragment (16x32 f16, M x K). Lane<16: row m0+lane, K {k0..k0+7, k0+16..k0+23}
//                               Lane>=16: row m0+lane-16, K {k0+8..15, k0+24..31}
DEVFN v16h frag_a(const h16* base, int pitch, int m0, int k0, int lane) {
  const h16* p = base + (m0 + (lane & 15)) * pitch + k0 + ((lane >> 4) << 3);
  v8h lo = *(const v8h*)p;
  v8h hi = *(const v8h*)(p + 16);
  v16h r;
#pragma unroll
  for (int i = 0; i < 8; ++i) { r[i] = lo[i]; r[i + 8] = hi[i]; }
  return r;
}

// B fragment (32x16 f16, K x N) read from an [n][k] (transposed) array:
// lane n = n0 + (lane&15); K {k0+8h..+7, k0+16+8h..+7}, h = lane>>4 (matches frag_a).
DEVFN v16h frag_b(const h16* base, int pitch, int n0, int k0, int lane) {
  const h16* p = base + (n0 + (lane & 15)) * pitch + k0 + ((lane >> 4) << 3);
  v8h lo = *(const v8h*)p;
  v8h hi = *(const v8h*)(p + 16);
  v16h r;
#pragma unroll
  for (int i = 0; i < 8; ++i) { r[i] = lo[i]; r[i + 8] = hi[i]; }
  return r;
}

// Issue one 2D TDM load: 256 rows x 25 f32, row stride 1600 f32 -> compact LDS.
// D# packing per CDNA5 ISA sec 8.3/8.4 (group0: count/lds/global/type;
// group1: data_size, dims, tile dims, strides). Groups 2/3 unused (2D tensor).
DEVFN void tdm_load_2d(const float* gsrc, unsigned lds_byte_off) {
  uint64_t ga = (uint64_t)(uintptr_t)gsrc;
  u32x4 g0;
  g0[0] = 1u;                                   // count=1, user mode, no gather
  g0[1] = lds_byte_off;                         // lds_addr (bytes)
  g0[2] = (unsigned)(ga & 0xFFFFFFFFu);         // global_addr[31:0]
  g0[3] = (unsigned)((ga >> 32) & 0x1FFFFFFu)   // global_addr[56:32]
          | (2u << 30);                         // type = 2 (image)
  i32x8 g1;
  g1[0] = 0x00020000;                           // data_size=2 (4B), mask=0, no flags
  g1[1] = (int)(cfg::V) << 16;                  // tensor_dim0 = 25   (bits 79:48)
  g1[2] = (int)(cfg::C) << 16;                  // tensor_dim1 = 256  (bits 111:80)
  g1[3] = (int)(cfg::V) << 16;                  // tile_dim0 = 25     (bits 127:112)
  g1[4] = (int)(cfg::C);                        // tile_dim1 = 256, tile_dim2 = 0
  g1[5] = (int)(cfg::TV);                       // tensor_dim0_stride = 1600
  g1[6] = 0;                                    // dim0_stride hi, dim1_stride lo
  g1[7] = 0;                                    // dim1_stride hi
  i32x4 gz = {0, 0, 0, 0};
  i32x8 gz8 = {0, 0, 0, 0, 0, 0, 0, 0};
  __builtin_amdgcn_tensor_load_to_lds(g0, g1, gz, gz, gz8, 0);
}

// ---------------------------------------------------------------------------
// Prep: convert weights to f16, transposed to [n][k] for B-fragment loads.
// ---------------------------------------------------------------------------
extern "C" __global__ __launch_bounds__(256)
void prep_weights(const float* __restrict__ Wqkv, const float* __restrict__ Wout,
                  h16* __restrict__ wqkv_t, h16* __restrict__ wout_t) {
  int idx = blockIdx.x * 256 + threadIdx.x;
  if (idx < cfg::C3 * cfg::C) {
    int n = idx / cfg::C, k = idx % cfg::C;
    wqkv_t[idx] = (h16)Wqkv[k * cfg::C3 + n];
  }
  if (idx < cfg::C * cfg::C) {
    int n = idx / cfg::C, k = idx % cfg::C;
    wout_t[idx] = (h16)Wout[k * cfg::C + n];
  }
}

// ---------------------------------------------------------------------------
// Fused cross-attention + out-proj + LayerNorm, one (b,t) per workgroup.
// ---------------------------------------------------------------------------
extern "C" __global__ __launch_bounds__(256)
void fused_attn_kernel(const float* __restrict__ feat1, const float* __restrict__ feat2,
                       const h16* __restrict__ wqkv_t, const h16* __restrict__ wout_t,
                       const float* __restrict__ b_out, const float* __restrict__ gamma,
                       const float* __restrict__ beta, float* __restrict__ out) {
  extern __shared__ char smem[];
  h16*   Xt0  = (h16*)(smem + cfg::OFF_X0);
  h16*   Xt1  = (h16*)(smem + cfg::OFF_X1);
  h16*   qkvB = (h16*)(smem + cfg::OFF_QKV);   // tile(br,part,h): +((br*3+part)*8+h)*1280 halves
  h16*   Pbuf = (h16*)(smem + cfg::OFF_P);
  h16*   Fus  = (h16*)(smem + cfg::OFF_F);
  float* Ybuf = (float*)(smem + cfg::OFF_Y);
  float* Stg0 = (float*)(smem + cfg::OFF_STG);
  float* Stg1 = (float*)(smem + cfg::OFF_STG + cfg::SZ_STG);

  const int tid  = threadIdx.x;
  const int lane = tid & 31;
  // wave id is uniform: force it into an SGPR so ownership tests become scalar
  // branches (keeps EXEC all-ones around WMMA, per ISA requirement).
  const int wave = __builtin_amdgcn_readfirstlane(tid >> 5);
  const int bt = blockIdx.x;
  const int b  = bt / cfg::T;
  const int t  = bt % cfg::T;

  // hint the weight tiles into cache early (global_prefetch_b8 path)
  __builtin_prefetch(wqkv_t + (size_t)wave * 8192, 0, 0);

  // ---- Phase 1a: TDM DMA both activation tiles into LDS staging ([c][v] f32) --
  const size_t src_off = (size_t)b * cfg::C * cfg::TV + (size_t)t * cfg::V;
  if (wave == 0) {
    tdm_load_2d(feat1 + src_off, (unsigned)cfg::OFF_STG);
    tdm_load_2d(feat2 + src_off, (unsigned)(cfg::OFF_STG + cfg::SZ_STG));
    __builtin_amdgcn_s_wait_tensorcnt(0);
  }
  __syncthreads();

  // ---- Phase 1b: transpose + convert staging -> f16 X tiles [row=v][k=c] ----
#pragma unroll
  for (int br = 0; br < 2; ++br) {
    const float* stg = br ? Stg1 : Stg0;
    h16* dst = br ? Xt1 : Xt0;
    for (int k = 0; k < 32; ++k) {
      int c = wave * 32 + k;
      float val = (lane < cfg::V) ? stg[c * cfg::V + lane] : 0.0f;
      dst[lane * cfg::XP + c] = (h16)val;   // rows 25..31 zero padded
    }
  }
  __syncthreads();

  // ---- Phase 2: QKV GEMM: [32x256] @ [256x768], both branches ----
  {
    int tcount = 0;
    for (int br = 0; br < 2; ++br)
      for (int mt = 0; mt < 2; ++mt)
        for (int nt = 0; nt < 48; ++nt) {
          if (((tcount++) & 7) != wave) continue;
          int n0 = nt * 16;
          const h16* xs = br ? Xt1 : Xt0;
          // preload all K-step B fragments so the compiler can clause the
          // global loads and stagger the waits across the 8 WMMAs
          v16h bfr[8];
#pragma unroll
          for (int kk = 0; kk < 8; ++kk)
            bfr[kk] = frag_b(wqkv_t, cfg::C, n0, kk * 32, lane);
          v8f acc = {};
#pragma unroll
          for (int kk = 0; kk < 8; ++kk) {
            v16h a = frag_a(xs, cfg::XP, mt * 16, kk * 32, lane);
            acc = wmma_f16(a, bfr[kk], acc);
          }
          int part = n0 >> 8;             // 0=q 1=k 2=v
          int h    = (n0 & 255) >> 5;     // head
          int dlo  = n0 & 31;             // 0 or 16
          h16* tile = qkvB + ((br * 3 + part) * 8 + h) * (cfg::SZ_TILE / 2);
          int col = dlo + (lane & 15);
          bool vpart = (part == 2);
#pragma unroll
          for (int r = 0; r < 8; ++r) {
            int row = mt * 16 + r + ((lane >> 4) << 3);
            // branchless: V stored transposed [d][j]; Q,K stored [joint][d]
            int addr = vpart ? (col * cfg::QP + row) : (row * cfg::QP + col);
            tile[addr] = (h16)acc[r];
          }
        }
  }
  __syncthreads();

  // zero the never-written pad rows (25..31) of the Fused tile
  for (int idx = tid; idx < 7 * cfg::XP; idx += 256)
    Fus[(25 + idx / cfg::XP) * cfg::XP + (idx % cfg::XP)] = (h16)0.0f;

  // ---- Phase 3: cross attention, one head per wave, both branches summed ----
  {
    const int h = wave;
    const float scale = 0.17677669529663687f;   // 1/sqrt(32)
    const int j0 = lane & 15;
    const bool hi_ok = (16 + j0) < cfg::V;       // mask cols 25..31
    v8f oacc[2][2] = {};
#pragma unroll
    for (int br = 0; br < 2; ++br) {
      h16* qt = qkvB + ((br * 3 + 0) * 8 + h) * (cfg::SZ_TILE / 2);
      h16* kt = qkvB + (((1 - br) * 3 + 1) * 8 + h) * (cfg::SZ_TILE / 2);
      h16* vt = qkvB + (((1 - br) * 3 + 2) * 8 + h) * (cfg::SZ_TILE / 2);
      v8f s[2][2];
#pragma unroll
      for (int mt = 0; mt < 2; ++mt)
#pragma unroll
        for (int nt = 0; nt < 2; ++nt) {
          v8f z = {};
          s[mt][nt] = wmma_f16(frag_a(qt, cfg::QP, mt * 16, 0, lane),
                               frag_b(kt, cfg::QP, nt * 16, 0, lane), z);
        }
      // row softmax across the two N-tiles; C-layout: vgpr r -> row r(+8), lane&15 -> col
      h16* pb = Pbuf + wave * (cfg::SZ_TILE / 2);
#pragma unroll
      for (int mt = 0; mt < 2; ++mt)
#pragma unroll
        for (int r = 0; r < 8; ++r) {
          float a0 = s[mt][0][r] * scale;
          float a1 = hi_ok ? s[mt][1][r] * scale : -3.0e38f;
          float m = fmaxf(a0, a1);
#pragma unroll
          for (int off = 8; off >= 1; off >>= 1) m = fmaxf(m, __shfl_xor(m, off, 16));
          float e0 = __expf(a0 - m);
          float e1 = hi_ok ? __expf(a1 - m) : 0.0f;
          float sum = e0 + e1;
#pragma unroll
          for (int off = 8; off >= 1; off >>= 1) sum += __shfl_xor(sum, off, 16);
          float inv = 1.0f / sum;
          int row = mt * 16 + r + ((lane >> 4) << 3);
          pb[row * cfg::QP + j0]      = (h16)(e0 * inv);
          pb[row * cfg::QP + 16 + j0] = (h16)(e1 * inv);
        }
      asm volatile("s_wait_dscnt 0" ::: "memory");   // in-wave LDS RAW on P tile
      // O += P @ V  (K = 32 joints, one WMMA step)
#pragma unroll
      for (int mt = 0; mt < 2; ++mt) {
        v16h a = frag_a(pb, cfg::QP, mt * 16, 0, lane);
#pragma unroll
        for (int nt = 0; nt < 2; ++nt)
          oacc[mt][nt] = wmma_f16(a, frag_b(vt, cfg::QP, nt * 16, 0, lane), oacc[mt][nt]);
      }
    }
    // scatter into Fused with the reference's head-mixing reshape:
    // flat f = h*(V*D) + i*D + d  ->  (row f/256, col f%256)
#pragma unroll
    for (int mt = 0; mt < 2; ++mt)
#pragma unroll
      for (int nt = 0; nt < 2; ++nt)
#pragma unroll
        for (int r = 0; r < 8; ++r) {
          int i = mt * 16 + r + ((lane >> 4) << 3);
          int d = nt * 16 + (lane & 15);
          if (i < cfg::V) {
            int f = h * (cfg::V * cfg::D) + i * cfg::D + d;
            Fus[(f >> 8) * cfg::XP + (f & 255)] = (h16)oacc[mt][nt][r];
          }
        }
  }
  __syncthreads();

  // ---- Phase 4: out projection [32x256] @ [256x256]; wave w owns cols w*32..+32 ----
  {
    v8f yacc[2][2] = {};
#pragma unroll
    for (int kk = 0; kk < 8; ++kk) {
      v16h a0 = frag_a(Fus, cfg::XP, 0, kk * 32, lane);
      v16h a1 = frag_a(Fus, cfg::XP, 16, kk * 32, lane);
#pragma unroll
      for (int nt = 0; nt < 2; ++nt) {
        v16h bb = frag_b(wout_t, cfg::C, wave * 32 + nt * 16, kk * 32, lane);
        yacc[0][nt] = wmma_f16(a0, bb, yacc[0][nt]);
        yacc[1][nt] = wmma_f16(a1, bb, yacc[1][nt]);
      }
    }
#pragma unroll
    for (int mt = 0; mt < 2; ++mt)
#pragma unroll
      for (int nt = 0; nt < 2; ++nt) {
        int n = wave * 32 + nt * 16 + (lane & 15);
        float bo = 2.0f * b_out[n];
#pragma unroll
        for (int r = 0; r < 8; ++r) {
          int row = mt * 16 + r + ((lane >> 4) << 3);
          Ybuf[row * cfg::YP + n] = yacc[mt][nt][r] + bo;
        }
      }
  }
  __syncthreads();

  // ---- Phase 5: LayerNorm over C per valid row, then transposed coalesced store ----
  for (int r = wave; r < cfg::V; r += 8) {
    float vals[8], s1 = 0.0f, s2 = 0.0f;
#pragma unroll
    for (int e = 0; e < 8; ++e) {
      float y = Ybuf[r * cfg::YP + lane * 8 + e];
      vals[e] = y; s1 += y; s2 += y * y;
    }
#pragma unroll
    for (int off = 16; off >= 1; off >>= 1) {
      s1 += __shfl_xor(s1, off, 32);
      s2 += __shfl_xor(s2, off, 32);
    }
    float mu = s1 * (1.0f / cfg::C);
    float var = s2 * (1.0f / cfg::C) - mu * mu;     // biased, matches jnp.var
    float rstd = rsqrtf(var + 1e-5f);
#pragma unroll
    for (int e = 0; e < 8; ++e) {
      int c = lane * 8 + e;
      Ybuf[r * cfg::YP + c] = (vals[e] - mu) * rstd * gamma[c] + beta[c];
    }
  }
  __syncthreads();
  float* obase = out + (size_t)b * cfg::C * cfg::TV + (size_t)t * cfg::V;
  for (int k = 0; k < 32; ++k) {
    int c = wave * 32 + k;
    if (lane < cfg::V)
      obase[(size_t)c * cfg::TV + lane] = Ybuf[lane * cfg::YP + c];
  }
}

// ---------------------------------------------------------------------------
extern "C" void kernel_launch(void* const* d_in, const int* in_sizes, int n_in,
                              void* d_out, int out_size, void* d_ws, size_t ws_size,
                              hipStream_t stream) {
  (void)in_sizes; (void)n_in; (void)out_size; (void)ws_size;
  const float* feat1 = (const float*)d_in[0];
  const float* feat2 = (const float*)d_in[1];
  const float* Wqkv  = (const float*)d_in[2];
  const float* Wout  = (const float*)d_in[3];
  const float* b_out = (const float*)d_in[4];
  const float* gamma = (const float*)d_in[5];
  const float* beta  = (const float*)d_in[6];
  float* out = (float*)d_out;

  h16* wqkv_t = (h16*)d_ws;                                         // 768x256 f16
  h16* wout_t = (h16*)((char*)d_ws + (size_t)cfg::C3 * cfg::C * 2); // 256x256 f16

  prep_weights<<<dim3((cfg::C3 * cfg::C + 255) / 256), dim3(256), 0, stream>>>(
      Wqkv, Wout, wqkv_t, wout_t);

  (void)hipFuncSetAttribute(reinterpret_cast<const void*>(fused_attn_kernel),
                            hipFuncAttributeMaxDynamicSharedMemorySize, cfg::SMEM);
  fused_attn_kernel<<<dim3(cfg::B * cfg::T), dim3(256), cfg::SMEM, stream>>>(
      feat1, feat2, wqkv_t, wout_t, b_out, gamma, beta, out);
}
